// FLAME_901943132660
// MI455X (gfx1250) — matrix operand order repaired and checked by reading, not compiled, not run on hardware
//
#include <hip/hip_runtime.h>
#include <math.h>

// ===========================================================================
// FLAME forward (vertices + per-vertex transforms) for MI455X / gfx1250.
//
// Sizes: V=5023, NJ=5, NSHAPE=100, NEXP=50, B=1024.
// d_out = [vertices (B*V*3 f32)] ++ [T (B*V*16 f32)]  (~391 MB -> HBM bound).
//
// Pipeline (all on `stream`, serialized by stream order):
//   P1  pack combined dirs matrix D[160, 3V] (shapedirs ++ jaw posedirs rows)
//       into WMMA B-fragment layout, f16.                 (~4.8 MB ws)
//   P2  fold J_regressor into shapedirs/v_template: Jsd[150,15], Jt[15].
//   KB  per-batch: betas->A-fragment layout (f16), rodrigues, joints,
//       kinematic chain, rel_transforms[b][5][4][4] (f32).
//   G   fused GEMM (v_wmma_f32_16x16x32_f16, K=160 in 5 steps) + fp32
//       v_template add + LBS epilogue through LDS, writing d_out directly.
// ===========================================================================

typedef __attribute__((ext_vector_type(16))) _Float16 v16h;
typedef __attribute__((ext_vector_type(8)))  float    v8f;
typedef __attribute__((ext_vector_type(4)))  float    v4f;

#define V_N    5023
#define B_N    1024
#define NCOL   15069      // 3*V
#define NCOLP  15072      // padded: multiple of 48 (and 16)
#define NT16   942        // NCOLP/16 WMMA column tiles
#define NGRP   314        // NCOLP/48 vertex groups (16 vertices each)
#define MT     64         // B/16 row tiles

// ---- workspace layout (bytes) ----
// relT : f32 [1024][80]        @ 0        (327680)
// Jsd  : f32 [150][16]         @ 327680   (9600)
// Jt   : f32 [16]              @ 337280   (64)
// Aswz : f16 [64][5][32][16]   @ 337344   (327680)   A fragments
// Bswz : f16 [942][5][32][16]  @ 665024   (4823040)  B fragments
// total 5,488,064 B
#define WS_RELT 0
#define WS_JSD  327680
#define WS_JT   337280
#define WS_ASWZ 337344
#define WS_BSWZ 665024

// ---------------------------------------------------------------------------
// P1: pack D[k, i] (k in [0,160), i in [0,NCOLP)) into B-fragment layout.
//   k < 150        : shapedirs[(v*3+c)*150 + k]  (i == v*3+c)
//   150 <= k < 159 : posedirs[(9 + k-150)*NCOL + i]   (jaw rows only)
//   else           : 0 (pad)
// B-frag (16-bit, 32x16): lane half h, elem e<8 -> K=8h+e, e>=8 ->
// K=16+8h+(e-8); n = lane&15.  One thread per (i, kstep) writes two
// 32B runs (one per lane-half).
// ---------------------------------------------------------------------------
__global__ __launch_bounds__(256) void flame_pack_B(
    const float* __restrict__ shapedirs, const float* __restrict__ posedirs,
    _Float16* __restrict__ Bswz)
{
  int t = blockIdx.x * 256 + threadIdx.x;
  if (t >= NCOLP * 5) return;
  int i = t % NCOLP;
  int kstep = t / NCOLP;
  int ntile = i >> 4, n = i & 15;
  #pragma unroll
  for (int half = 0; half < 2; ++half) {
    v16h hv;
    #pragma unroll
    for (int e = 0; e < 16; ++e) {
      int k = kstep * 32 + ((e < 8) ? (half * 8 + e) : (16 + half * 8 + (e - 8)));
      float val = 0.0f;
      if (i < NCOL) {
        if (k < 150)      val = shapedirs[(size_t)i * 150 + k];
        else if (k < 159) val = posedirs[(size_t)(k - 141) * NCOL + i];
      }
      hv[e] = (_Float16)val;
    }
    *(v16h*)(Bswz + ((size_t)(ntile * 5 + kstep) * 32 + (n + 16 * half)) * 16) = hv;
  }
}

// ---------------------------------------------------------------------------
// P2: Jsd[l][jc] = sum_v Jreg[j,v] * shapedirs[v,c,l]   (l < 150)
//     Jt[jc]     = sum_v Jreg[j,v] * v_template[v,c]    (block l == 150)
// One block per (l, jc); 256-thread LDS reduction over V.
// ---------------------------------------------------------------------------
__global__ __launch_bounds__(256) void flame_fold_J(
    const float* __restrict__ Jreg, const float* __restrict__ shapedirs,
    const float* __restrict__ vtempl, float* __restrict__ Jsd,
    float* __restrict__ Jt)
{
  __shared__ float red[256];
  int idx = blockIdx.x;                  // 0 .. 151*15-1
  int l = idx / 15, jc = idx % 15;
  int j = jc / 3, c = jc % 3;
  float s = 0.0f;
  for (int v = threadIdx.x; v < V_N; v += 256) {
    float x = (l < 150) ? shapedirs[((size_t)v * 3 + c) * 150 + l]
                        : vtempl[v * 3 + c];
    s += Jreg[(size_t)j * V_N + v] * x;
  }
  red[threadIdx.x] = s;
  __syncthreads();
  for (int off = 128; off > 0; off >>= 1) {
    if (threadIdx.x < off) red[threadIdx.x] += red[threadIdx.x + off];
    __syncthreads();
  }
  if (threadIdx.x == 0) {
    if (l < 150) Jsd[l * 16 + jc] = red[0];
    else         Jt[jc] = red[0];
  }
}

// ---------------------------------------------------------------------------
// KB: per-batch rigid math + A-fragment packing
// ---------------------------------------------------------------------------
__device__ __forceinline__ void rodrigues9(float rx, float ry, float rz,
                                           float* __restrict__ R)
{
  float ex = rx + 1e-8f, ey = ry + 1e-8f, ez = rz + 1e-8f;
  float ang = sqrtf(ex * ex + ey * ey + ez * ez);
  float inv = 1.0f / ang;
  float ax = rx * inv, ay = ry * inv, az = rz * inv;
  float s = sinf(ang), c = cosf(ang), oc = 1.0f - c;
  float K[9] = {0.0f, -az, ay,  az, 0.0f, -ax,  -ay, ax, 0.0f};
  float K2[9];
  #pragma unroll
  for (int r = 0; r < 3; ++r)
    #pragma unroll
    for (int cc = 0; cc < 3; ++cc)
      K2[r * 3 + cc] = K[r * 3 + 0] * K[0 + cc] + K[r * 3 + 1] * K[3 + cc] +
                       K[r * 3 + 2] * K[6 + cc];
  #pragma unroll
  for (int e = 0; e < 9; ++e)
    R[e] = ((e == 0 || e == 4 || e == 8) ? 1.0f : 0.0f) + s * K[e] + oc * K2[e];
}

__device__ __forceinline__ void mat3mul(const float* A, const float* B, float* C)
{
  #pragma unroll
  for (int r = 0; r < 3; ++r)
    #pragma unroll
    for (int cc = 0; cc < 3; ++cc)
      C[r * 3 + cc] = A[r * 3 + 0] * B[0 + cc] + A[r * 3 + 1] * B[3 + cc] +
                      A[r * 3 + 2] * B[6 + cc];
}

__device__ __forceinline__ void mat3vec(const float* A, const float* v, float* o)
{
  #pragma unroll
  for (int r = 0; r < 3; ++r)
    o[r] = A[r * 3 + 0] * v[0] + A[r * 3 + 1] * v[1] + A[r * 3 + 2] * v[2];
}

__global__ __launch_bounds__(256) void flame_batch(
    const float* __restrict__ shape, const float* __restrict__ expr,
    const float* __restrict__ pose, const float* __restrict__ Jsd,
    const float* __restrict__ Jt, float* __restrict__ relT,
    _Float16* __restrict__ Aswz)
{
  int b = blockIdx.x * 256 + threadIdx.x;
  if (b >= B_N) return;

  // joints[jc] = Jt[jc] + sum_l betas[l] * Jsd[l][jc]
  float jp[15];
  #pragma unroll
  for (int jc = 0; jc < 15; ++jc) jp[jc] = Jt[jc];
  for (int l = 0; l < 150; ++l) {
    float bl = (l < 100) ? shape[b * 100 + l] : expr[b * 50 + (l - 100)];
    #pragma unroll
    for (int jc = 0; jc < 15; ++jc) jp[jc] += bl * Jsd[l * 16 + jc];
  }

  float R0[9], R2[9];
  rodrigues9(pose[b * 6 + 0], pose[b * 6 + 1], pose[b * 6 + 2], R0);  // global
  rodrigues9(pose[b * 6 + 3], pose[b * 6 + 4], pose[b * 6 + 5], R2);  // jaw

  // kinematic chain (PARENTS = [-1,0,1,1,1]); joints 1,3,4 are identity rot.
  float RG[5][9], tG[5][3];
  #pragma unroll
  for (int e = 0; e < 9; ++e) {
    RG[0][e] = R0[e]; RG[1][e] = R0[e]; RG[3][e] = R0[e]; RG[4][e] = R0[e];
  }
  mat3mul(R0, R2, RG[2]);
  #pragma unroll
  for (int c = 0; c < 3; ++c) tG[0][c] = jp[c];
  float d1[3] = {jp[3] - jp[0], jp[4] - jp[1], jp[5] - jp[2]};
  float t1[3]; mat3vec(R0, d1, t1);
  #pragma unroll
  for (int c = 0; c < 3; ++c) tG[1][c] = t1[c] + tG[0][c];
  #pragma unroll
  for (int jj = 2; jj < 5; ++jj) {
    float dd[3] = {jp[jj * 3 + 0] - jp[3], jp[jj * 3 + 1] - jp[4],
                   jp[jj * 3 + 2] - jp[5]};
    float tt[3]; mat3vec(R0, dd, tt);
    #pragma unroll
    for (int c = 0; c < 3; ++c) tG[jj][c] = tt[c] + tG[1][c];
  }

  // rel_transforms: [RG | tG - RG*joint ; 0 0 0 1]
  #pragma unroll
  for (int jj = 0; jj < 5; ++jj) {
    float rj[3]; mat3vec(RG[jj], &jp[jj * 3], rj);
    float* o = relT + (size_t)b * 80 + jj * 16;
    #pragma unroll
    for (int r = 0; r < 3; ++r) {
      o[r * 4 + 0] = RG[jj][r * 3 + 0];
      o[r * 4 + 1] = RG[jj][r * 3 + 1];
      o[r * 4 + 2] = RG[jj][r * 3 + 2];
      o[r * 4 + 3] = tG[jj][r] - rj[r];
    }
    o[12] = 0.0f; o[13] = 0.0f; o[14] = 0.0f; o[15] = 1.0f;
  }

  // jaw pose feature (rows 9..17 of pose_feature)
  float feat[9];
  #pragma unroll
  for (int e = 0; e < 9; ++e)
    feat[e] = R2[e] - ((e == 0 || e == 4 || e == 8) ? 1.0f : 0.0f);

  // A-fragment pack: 16-bit A 16x32 layout: lane = m + 16*half,
  //   elem e<8  -> K = kstep*32 + half*8 + e
  //   elem e>=8 -> K = kstep*32 + 16 + half*8 + (e-8)
  int mtile = b >> 4, m = b & 15;
  #pragma unroll
  for (int ks = 0; ks < 5; ++ks) {
    #pragma unroll
    for (int half = 0; half < 2; ++half) {
      v16h hv;
      #pragma unroll
      for (int e = 0; e < 16; ++e) {
        int k = ks * 32 + ((e < 8) ? (half * 8 + e) : (16 + half * 8 + (e - 8)));
        float val;
        if (k < 100)      val = shape[b * 100 + k];
        else if (k < 150) val = expr[b * 50 + (k - 100)];
        else if (k < 159) val = feat[k - 150];
        else              val = 0.0f;
        hv[e] = (_Float16)val;
      }
      *(v16h*)(Aswz + ((size_t)(mtile * 5 + ks) * 32 + (m + 16 * half)) * 16) = hv;
    }
  }
}

// ---------------------------------------------------------------------------
// G: fused WMMA GEMM + LBS epilogue.
// Block = 256 threads = 8 waves; blockIdx.y = mtile (16 batch rows);
// each wave owns one 48-column group (16 vertices): 3 subtiles x 5 WMMA.
// Results staged in wave-private LDS (same-wave LDS is in-order), then
// LBS: T = sum_j w[v,j]*rel[b,j], vertex = T * [p;1]; write straight to out.
// ---------------------------------------------------------------------------
__global__ __launch_bounds__(256) void flame_gemm_lbs(
    const _Float16* __restrict__ Aswz, const _Float16* __restrict__ Bswz,
    const float* __restrict__ relT, const float* __restrict__ vtempl,
    const float* __restrict__ lbsw, float* __restrict__ outV,
    float* __restrict__ outT)
{
  __shared__ float relS[16 * 80];       // rel transforms for this mtile
  __shared__ float tileS[8 * 960];      // per-wave 48 cols x 16 rows, stride 20

  int mtile = blockIdx.y;
  int tid = threadIdx.x;
  for (int k = tid; k < 16 * 80; k += 256)
    relS[k] = relT[(size_t)(mtile * 16 + k / 80) * 80 + (k % 80)];
  __syncthreads();

  int w = tid >> 5, lane = tid & 31;
  int g = blockIdx.x * 8 + w;           // vertex group (wave-uniform)
  if (g < NGRP) {
    const v16h* Ap = (const v16h*)Aswz;
    const v16h* Bp = (const v16h*)Bswz;
    v16h a[5];
    #pragma unroll
    for (int ks = 0; ks < 5; ++ks)
      a[ks] = Ap[(size_t)(mtile * 5 + ks) * 32 + lane];

    float* myTile = &tileS[w * 960];
    int n = lane & 15, hi = lane >> 4;

    #pragma unroll
    for (int t = 0; t < 3; ++t) {
      int ntile = g * 3 + t;
      v8f acc = {};
      #pragma unroll
      for (int ks = 0; ks < 5; ++ks) {
        v16h bf = Bp[(size_t)(ntile * 5 + ks) * 32 + lane];
        acc = __builtin_amdgcn_wmma_f32_16x16x32_f16(
            false, a[ks], false, bf, (short)0, acc, false, false);
      }
      // fp32 v_template add (same column for all 8 rows of this lane)
      int ci = ntile * 16 + n;
      float vt = (ci < NCOL) ? vtempl[ci] : 0.0f;
      // C/D layout: VGPR r -> row (r + 8*hi), col = n.  Store [col][row].
      float* dst = &myTile[(t * 16 + n) * 20 + hi * 8];
      v4f lo = {acc[0] + vt, acc[1] + vt, acc[2] + vt, acc[3] + vt};
      v4f hh = {acc[4] + vt, acc[5] + vt, acc[6] + vt, acc[7] + vt};
      *(v4f*)dst = lo;
      *(v4f*)(dst + 4) = hh;
    }

    // LBS over 16 rows x 16 vertices = 256 pairs, 8 per lane.
    #pragma unroll
    for (int it = 0; it < 8; ++it) {
      int m = it * 2 + hi;              // batch row in tile
      int vloc = n;
      int v = g * 16 + vloc;
      if (v < V_N) {
        float px = myTile[(3 * vloc + 0) * 20 + m];
        float py = myTile[(3 * vloc + 1) * 20 + m];
        float pz = myTile[(3 * vloc + 2) * 20 + m];
        float w0 = lbsw[(size_t)v * 5 + 0];
        float w1 = lbsw[(size_t)v * 5 + 1];
        float w2 = lbsw[(size_t)v * 5 + 2];
        float w3 = lbsw[(size_t)v * 5 + 3];
        float w4 = lbsw[(size_t)v * 5 + 4];
        const float* rm = &relS[m * 80];
        float T[16];
        #pragma unroll
        for (int x = 0; x < 16; ++x)
          T[x] = w0 * rm[x] + w1 * rm[16 + x] + w2 * rm[32 + x] +
                 w3 * rm[48 + x] + w4 * rm[64 + x];
        int b = mtile * 16 + m;
        size_t base = (size_t)b * V_N + v;
        outV[base * 3 + 0] = T[0] * px + T[1] * py + T[2] * pz + T[3];
        outV[base * 3 + 1] = T[4] * px + T[5] * py + T[6] * pz + T[7];
        outV[base * 3 + 2] = T[8] * px + T[9] * py + T[10] * pz + T[11];
        float* to = outT + base * 16;
        *(v4f*)(to + 0)  = (v4f){T[0], T[1], T[2], T[3]};
        *(v4f*)(to + 4)  = (v4f){T[4], T[5], T[6], T[7]};
        *(v4f*)(to + 8)  = (v4f){T[8], T[9], T[10], T[11]};
        *(v4f*)(to + 12) = (v4f){T[12], T[13], T[14], T[15]};
      }
    }
  }
}

// ---------------------------------------------------------------------------
extern "C" void kernel_launch(void* const* d_in, const int* in_sizes, int n_in,
                              void* d_out, int out_size, void* d_ws,
                              size_t ws_size, hipStream_t stream)
{
  const float* shape  = (const float*)d_in[0];  // [B,100]
  const float* expr   = (const float*)d_in[1];  // [B,50]
  const float* pose   = (const float*)d_in[2];  // [B,6]
  const float* vtempl = (const float*)d_in[3];  // [V,3]
  const float* sdirs  = (const float*)d_in[4];  // [V,3,150]
  const float* pdirs  = (const float*)d_in[5];  // [36, V*3]
  const float* Jreg   = (const float*)d_in[6];  // [5, V]
  const float* lbsw   = (const float*)d_in[7];  // [V, 5]

  char* ws = (char*)d_ws;
  float*    relT = (float*)(ws + WS_RELT);
  float*    Jsd  = (float*)(ws + WS_JSD);
  float*    Jt   = (float*)(ws + WS_JT);
  _Float16* Aswz = (_Float16*)(ws + WS_ASWZ);
  _Float16* Bswz = (_Float16*)(ws + WS_BSWZ);

  float* outV = (float*)d_out;
  float* outT = outV + (size_t)B_N * V_N * 3;

  flame_pack_B<<<(NCOLP * 5 + 255) / 256, 256, 0, stream>>>(sdirs, pdirs, Bswz);
  flame_fold_J<<<151 * 15, 256, 0, stream>>>(Jreg, sdirs, vtempl, Jsd, Jt);
  flame_batch<<<(B_N + 255) / 256, 256, 0, stream>>>(shape, expr, pose, Jsd,
                                                     Jt, relT, Aswz);
  flame_gemm_lbs<<<dim3((NGRP + 7) / 8, MT), 256, 0, stream>>>(
      Aswz, Bswz, relT, vtempl, lbsw, outV, outT);
}
